// PriorBernoulliSTBP_2181843386748
// MI455X (gfx1250) — compile-verified, matrix operand order checked by reading, not run on hardware
//
#include <hip/hip_runtime.h>

typedef __attribute__((ext_vector_type(16))) __bf16 v16bf;
typedef __attribute__((ext_vector_type(8)))  float  v8f;

#define TT 16
#define BB 8192
#define K1P 64    // layer1 K padded (56 -> 64)
#define K2P 128   // layer2 K padded (112 -> 128)
#define K3  224   // layer3 K exact
#define NT1 7     // 112/16
#define NT2 14    // 224/16
#define NT3 70    // 1120/16
#define KC1 2
#define KC2 4
#define KC3 7
#define C3  1120

__device__ __forceinline__ v8f vzero() {
  v8f r;
#pragma unroll
  for (int i = 0; i < 8; ++i) r[i] = 0.0f;
  return r;
}

__device__ __forceinline__ v8f wmma_bf16(v16bf a, v16bf b, v8f c) {
  return __builtin_amdgcn_wmma_f32_16x16x32_bf16(
      false, a, false, b, (short)0, c, false, false);
}

// CDNA5 async copy: global -> LDS, 16 bytes per lane, tracked by ASYNCcnt.
// IOFFSET is added to BOTH the LDS and global addresses (ISA 15.18.3).
__device__ __forceinline__ void async_b128(const void* gsrc, const __bf16* lds_dst) {
  uint32_t l = (uint32_t)(size_t)(__attribute__((address_space(3))) const __bf16*)lds_dst;
  uint64_t g = (uint64_t)(size_t)gsrc;
  asm volatile("global_load_async_to_lds_b128 %0, %1, off" :: "v"(l), "v"(g) : "memory");
}
__device__ __forceinline__ void async_b128_off3584(const void* gsrc, const __bf16* lds_dst) {
  uint32_t l = (uint32_t)(size_t)(__attribute__((address_space(3))) const __bf16*)lds_dst;
  uint64_t g = (uint64_t)(size_t)gsrc;
  asm volatile("global_load_async_to_lds_b128 %0, %1, off offset:3584" :: "v"(l), "v"(g) : "memory");
}
#define WAIT_ASYNC0() asm volatile("s_wait_asynccnt 0x0" ::: "memory")
#define WAIT_ASYNC2() asm volatile("s_wait_asynccnt 0x2" ::: "memory")
#define WAIT_DS0()    asm volatile("s_wait_dscnt 0x0"    ::: "memory")

// ---------------- prep: teacher-forced input, bf16, K padded to 64 ----------
__global__ void pack_z_kernel(const float* __restrict__ z, __bf16* __restrict__ zp) {
  int idx = blockIdx.x * blockDim.x + threadIdx.x;   // < 16*8192*64 = 2^23
  int c = idx & 63;
  int b = (idx >> 6) & 8191;
  int t = idx >> 19;
  float v = 0.0f;
  if (t > 0 && c < 56) v = z[((size_t)(t - 1) * BB + b) * 56 + c];
  zp[idx] = (__bf16)v;
}

// ---------------- prep: weights -> hi/lo bf16 in WMMA B-fragment layout -----
// frag (kc, nt): lane l holds column n = nt*16 + (l&15), K = kc*32 + (l>>4)*16 + i
__global__ void pack_w_kernel(const float* __restrict__ W, __bf16* __restrict__ hi,
                              __bf16* __restrict__ lo, int K, int NT, int total) {
  int idx = blockIdx.x * blockDim.x + threadIdx.x;
  if (idx >= total) return;
  int i  = idx & 15;
  int l  = (idx >> 4) & 31;
  int nt = (idx >> 9) % NT;
  int kc = idx / (NT << 9);
  int n = nt * 16 + (l & 15);
  int k = kc * 32 + ((l >> 4) << 4) + i;
  float w = (k < K) ? W[(size_t)n * K + k] : 0.0f;
  __bf16 h = (__bf16)w;
  float r = w - (float)h;
  hi[idx] = h;
  lo[idx] = (__bf16)r;
}

// ---------------- phase A: layers 1+2 -----------
// 2 waves per block, each wave owns 16 batch rows; W1/W2 hi/lo packs staged
// once into LDS via async copy (143,360 B contiguous in workspace).
__global__ __launch_bounds__(64) void snn_layers12(
    const __bf16* __restrict__ zp,
    const __bf16* __restrict__ wsrc,   // contiguous: w1hi|w1lo|w2hi|w2lo
    const float* __restrict__ b1, const float* __restrict__ b2,
    __bf16* __restrict__ s2g) {
  __shared__ __bf16 smem[75776];       // 151,552 B

  const int tid  = threadIdx.x;
  const int lane = tid & 31;
  const int wv   = tid >> 5;
  const int nlo  = lane & 15;
  const int hh   = (lane >> 4) & 1;
  const int brow = (blockIdx.x * 2 + wv) * 16;

  __bf16* s1buf = smem + wv * 2048;            // 16 x K2P per wave
  const __bf16* lw1hi = smem + 4096;           // 7,168 halfs
  const __bf16* lw1lo = smem + 11264;          // 7,168
  const __bf16* lw2hi = smem + 18432;          // 28,672
  const __bf16* lw2lo = smem + 47104;          // 28,672

  // stage weight fragments into LDS: 8960 x 16B chunks
  for (int c = tid; c < 8960; c += 64) {
    async_b128((const char*)wsrc + (size_t)c * 16, smem + 4096 + c * 8);
  }
  // zero own s1 staging buffer (K padding 112..127 must stay 0)
  {
    v16bf zv;
#pragma unroll
    for (int i = 0; i < 16; ++i) zv[i] = (__bf16)0.0f;
    for (int i = lane * 16; i < 2048; i += 32 * 16) *(v16bf*)(s1buf + i) = zv;
  }
  WAIT_ASYNC0();
  WAIT_DS0();
  __syncthreads();

  float b1v[NT1], b2v[NT2];
#pragma unroll
  for (int nt = 0; nt < NT1; ++nt) b1v[nt] = b1[nt * 16 + nlo];
#pragma unroll
  for (int nt = 0; nt < NT2; ++nt) b2v[nt] = b2[nt * 16 + nlo];

  v8f v1[NT1], v2[NT2];
#pragma unroll
  for (int nt = 0; nt < NT1; ++nt) v1[nt] = vzero();
#pragma unroll
  for (int nt = 0; nt < NT2; ++nt) v2[nt] = vzero();

  for (int t = 0; t < TT; ++t) {
    // ---- layer 1 (input read exactly once -> non-temporal loads) ----
    v16bf a1[KC1];
#pragma unroll
    for (int kc = 0; kc < KC1; ++kc) {
      const __bf16* p = zp + ((size_t)(t * BB + brow + nlo) * K1P + kc * 32 + hh * 16);
      a1[kc] = __builtin_nontemporal_load((const v16bf*)p);
    }
#pragma unroll
    for (int nt = 0; nt < NT1; ++nt) {
      v8f acc = vzero();
#pragma unroll
      for (int kc = 0; kc < KC1; ++kc) {
        const v16bf bh = *(const v16bf*)(lw1hi + ((kc * NT1 + nt) * 32 + lane) * 16);
        const v16bf bl = *(const v16bf*)(lw1lo + ((kc * NT1 + nt) * 32 + lane) * 16);
        acc = wmma_bf16(a1[kc], bh, acc);
        acc = wmma_bf16(a1[kc], bl, acc);
      }
#pragma unroll
      for (int i = 0; i < 8; ++i) {
        float h = 0.5f * (v1[nt][i] + acc[i] + b1v[nt]);  // tau=2, v_reset=0
        bool s = (h >= 1.0f);
        v1[nt][i] = s ? 0.0f : h;
        int m = i + hh * 8;
        s1buf[m * K2P + nt * 16 + nlo] = s ? (__bf16)1.0f : (__bf16)0.0f;
      }
    }
    WAIT_DS0();

    // ---- layer 2 ----
    v16bf a2[KC2];
#pragma unroll
    for (int kc = 0; kc < KC2; ++kc) {
      a2[kc] = *(const v16bf*)(s1buf + nlo * K2P + kc * 32 + hh * 16);
    }
#pragma unroll
    for (int nt = 0; nt < NT2; ++nt) {
      v8f acc = vzero();
#pragma unroll
      for (int kc = 0; kc < KC2; ++kc) {
        const v16bf bh = *(const v16bf*)(lw2hi + ((kc * NT2 + nt) * 32 + lane) * 16);
        const v16bf bl = *(const v16bf*)(lw2lo + ((kc * NT2 + nt) * 32 + lane) * 16);
        acc = wmma_bf16(a2[kc], bh, acc);
        acc = wmma_bf16(a2[kc], bl, acc);
      }
      // s2 is re-read 5x by phase B: keep RT (L2-resident) stores
#pragma unroll
      for (int i = 0; i < 8; ++i) {
        float h = 0.5f * (v2[nt][i] + acc[i] + b2v[nt]);
        bool s = (h >= 1.0f);
        v2[nt][i] = s ? 0.0f : h;
        int m = i + hh * 8;
        s2g[(size_t)(t * BB + brow + m) * K3 + nt * 16 + nlo] = s ? (__bf16)1.0f : (__bf16)0.0f;
      }
    }
  }
}

// ---------------- phase B: layer 3 -----------
// 7 waves per block; each wave owns 2 nt-tiles with W3 fragments resident in
// registers across the whole T loop. The shared 16x224 bf16 s2 tile (7168 B,
// contiguous) is double-buffered in LDS via async global->LDS loads.
__global__ __launch_bounds__(224) void snn_layer3(
    const __bf16* __restrict__ s2g,
    const __bf16* __restrict__ w3hi, const __bf16* __restrict__ w3lo,
    const float* __restrict__ b3, float* __restrict__ out) {
  __shared__ __bf16 tile[2][16 * K3];  // 2 x 7168 B

  const int tid  = threadIdx.x;        // 0..223
  const int lane = tid & 31;
  const int wv   = tid >> 5;           // 0..6
  const int nlo  = lane & 15;
  const int hh   = lane >> 4;
  const int brow = blockIdx.x * 16;
  const int nt0  = blockIdx.y * 14 + wv * 2;   // this wave's first nt tile

  // resident W3 fragments: 2 nt x 7 kc x {hi,lo} = 28 x v16bf
  v16bf wf[2][KC3][2];
#pragma unroll
  for (int j = 0; j < 2; ++j)
#pragma unroll
    for (int kc = 0; kc < KC3; ++kc) {
      wf[j][kc][0] = *(const v16bf*)(w3hi + (((size_t)kc * NT3 + nt0 + j) * 32 + lane) * 16);
      wf[j][kc][1] = *(const v16bf*)(w3lo + (((size_t)kc * NT3 + nt0 + j) * 32 + lane) * 16);
    }
  float b3v[2];
  b3v[0] = b3[(nt0 + 0) * 16 + nlo];
  b3v[1] = b3[(nt0 + 1) * 16 + nlo];

  v8f v3[2];
  v3[0] = vzero();
  v3[1] = vzero();

  // prefetch tile for t=0 (each thread moves 2 x 16B; 448 chunks total)
  {
    const __bf16* g = s2g + (size_t)brow * K3;
    async_b128((const char*)g + (size_t)tid * 16, &tile[0][0] + tid * 8);
    async_b128_off3584((const char*)g + (size_t)tid * 16, &tile[0][0] + tid * 8);
  }

  for (int t = 0; t < TT; ++t) {
    if (t + 1 < TT) {
      const __bf16* g = s2g + (size_t)((t + 1) * BB + brow) * K3;
      __bf16* l = &tile[(t + 1) & 1][0];
      async_b128((const char*)g + (size_t)tid * 16, l + tid * 8);
      async_b128_off3584((const char*)g + (size_t)tid * 16, l + tid * 8);
      WAIT_ASYNC2();                   // tile t complete, t+1 may be in flight
    } else {
      WAIT_ASYNC0();
    }
    __syncthreads();

    const __bf16* tb = &tile[t & 1][0];
    v8f acc0 = vzero(), acc1 = vzero();
#pragma unroll
    for (int kc = 0; kc < KC3; ++kc) {
      const v16bf a = *(const v16bf*)(tb + nlo * K3 + kc * 32 + hh * 16);
      acc0 = wmma_bf16(a, wf[0][kc][0], acc0);
      acc0 = wmma_bf16(a, wf[0][kc][1], acc0);
      acc1 = wmma_bf16(a, wf[1][kc][0], acc1);
      acc1 = wmma_bf16(a, wf[1][kc][1], acc1);
    }
#pragma unroll
    for (int j = 0; j < 2; ++j) {
      v8f* accp = (j == 0) ? &acc0 : &acc1;
#pragma unroll
      for (int i = 0; i < 8; ++i) {
        float h = 0.5f * (v3[j][i] + (*accp)[i] + b3v[j]);
        bool s = (h >= 1.0f);
        v3[j][i] = s ? 0.0f : h;
        int m = i + hh * 8;
        // 587 MB write-once stream: non-temporal so it doesn't sweep L2
        __builtin_nontemporal_store(
            s ? 1.0f : 0.0f,
            &out[(size_t)(t * BB + brow + m) * C3 + (nt0 + j) * 16 + nlo]);
      }
    }
    __syncthreads();                   // protect buffer reuse vs next prefetch
  }
}

// ---------------- launch ----------------------------------------------------
extern "C" void kernel_launch(void* const* d_in, const int* in_sizes, int n_in,
                              void* d_out, int out_size, void* d_ws, size_t ws_size,
                              hipStream_t stream) {
  const float* z  = (const float*)d_in[0];
  const float* W1 = (const float*)d_in[1];
  const float* b1 = (const float*)d_in[2];
  const float* W2 = (const float*)d_in[3];
  const float* b2 = (const float*)d_in[4];
  const float* W3 = (const float*)d_in[5];
  const float* b3 = (const float*)d_in[6];
  float* out = (float*)d_out;

  // workspace layout (bytes); weight packs are contiguous for the LDS stage
  uint8_t* ws = (uint8_t*)d_ws;
  __bf16* zp   = (__bf16*)(ws);                       // 16*8192*64  *2 = 16,777,216
  __bf16* s2g  = (__bf16*)(ws + 16777216);            // 16*8192*224 *2 = 58,720,256
  __bf16* w1hi = (__bf16*)(ws + 75497472);            // 7,168 elems
  __bf16* w1lo = (__bf16*)(ws + 75511808);            // 7,168
  __bf16* w2hi = (__bf16*)(ws + 75526144);            // 28,672
  __bf16* w2lo = (__bf16*)(ws + 75583488);            // 28,672
  __bf16* w3hi = (__bf16*)(ws + 75640832);            // 250,880
  __bf16* w3lo = (__bf16*)(ws + 76142592);            // end @ 76,644,352

  // prep
  pack_z_kernel<<<(16 * BB * K1P) / 256, 256, 0, stream>>>(z, zp);
  pack_w_kernel<<<(KC1 * NT1 * 512 + 255) / 256, 256, 0, stream>>>(W1, w1hi, w1lo, 56,  NT1, KC1 * NT1 * 512);
  pack_w_kernel<<<(KC2 * NT2 * 512 + 255) / 256, 256, 0, stream>>>(W2, w2hi, w2lo, 112, NT2, KC2 * NT2 * 512);
  pack_w_kernel<<<(KC3 * NT3 * 512 + 255) / 256, 256, 0, stream>>>(W3, w3hi, w3lo, 224, NT3, KC3 * NT3 * 512);

  // phase A: layers 1+2 over all T (2 waves/block, 16 batch rows per wave)
  snn_layers12<<<BB / 32, 64, 0, stream>>>(zp, w1hi, b1, b2, s2g);

  // phase B: layer 3 (512 batch tiles x 5 nt-groups, 7 waves/block)
  snn_layer3<<<dim3(BB / 16, 5), 224, 0, stream>>>(s2g, w3hi, w3lo, b3, out);
}
